// TnoBlock2d_481036337190
// MI455X (gfx1250) — matrix operand order repaired and checked by reading, not compiled, not run on hardware
//
#include <hip/hip_runtime.h>
#include <hip/hip_bf16.h>

typedef _Float16 half_t;
typedef __attribute__((ext_vector_type(16))) _Float16 v16h;
typedef __attribute__((ext_vector_type(8)))  _Float16 h8;
typedef __attribute__((ext_vector_type(8)))  float    v8f;

#define PI_F 3.14159265358979323846f

// =====================================================================
// Batched WMMA GEMM, all-aligned fast path (M%16==0, N%(16*NT)==0, K%32==0):
//   C[b] (MxN f32, ldc) (+)= A[b] (MxK f16, lda) * Bt[b]^T   (Bt stored [N,K], ldb)
// Templated on NT: one wave computes a 16x(16*NT) strip of C, reusing one A
// fragment over NT B fragments -> NT independent back-to-back
// v_wmma_f32_16x16x32_f16 per K step (NT=4 covers the WMMA->WMMA RAW hazard
// distance with useful work). No runtime conditionals in the loop: EXEC stays
// all-ones, fragments are pure 16-byte GLOBAL_LOAD_B128 pairs.
// Batch strides in elements; stride 0 => shared operand.
// =====================================================================
template <int NT>
__global__ void gemm16_k(const half_t* __restrict__ A, const half_t* __restrict__ Bt,
                         float* __restrict__ C, int M, int N, int K,
                         int lda, int ldb, int ldc,
                         long sA, long sB, long sC, int beta)
{
    const int lane    = threadIdx.x;                            // 0..31
    const int tilesM  = M >> 4;
    const int tilesNg = (N >> 4) / NT;                          // exact by contract
    const int wid     = blockIdx.x * blockDim.y + threadIdx.y;  // wave-uniform
    if (wid >= tilesM * tilesNg) return;                        // uniform exit
    const int tm  = wid / tilesNg;
    const int tn0 = (wid - tm * tilesNg) * NT;

    const long b = blockIdx.y;
    A += b * sA; Bt += b * sB; C += b * sC;

    const int g = lane >> 4;                                    // half-wave group
    const int r = lane & 15;
    const half_t* aRow = A + (long)(tm * 16 + r) * lda;
    const half_t* bRow[NT];
#pragma unroll
    for (int t = 0; t < NT; ++t)
        bRow[t] = Bt + (long)((tn0 + t) * 16 + r) * ldb;

    v8f acc[NT];
#pragma unroll
    for (int t = 0; t < NT; ++t) {
        const int col = (tn0 + t) * 16 + r;
#pragma unroll
        for (int j = 0; j < 8; ++j)
            acc[t][j] = beta ? C[(long)(tm * 16 + j + 8 * g) * ldc + col] : 0.0f;
    }

    for (int k0 = 0; k0 < K; k0 += 32) {
        h8 alo = *(const h8*)(aRow + k0 + 8 * g);               // K = {0..7}+8g
        h8 ahi = *(const h8*)(aRow + k0 + 16 + 8 * g);          // K = {16..23}+8g
        v16h af = __builtin_shufflevector(alo, ahi, 0,1,2,3,4,5,6,7,8,9,10,11,12,13,14,15);
#pragma unroll
        for (int t = 0; t < NT; ++t) {
            h8 blo = *(const h8*)(bRow[t] + k0 + 8 * g);        // K = {0..7}+8g
            h8 bhi = *(const h8*)(bRow[t] + k0 + 16 + 8 * g);   // K = {16..23}+8g
            v16h bf = __builtin_shufflevector(blo, bhi, 0,1,2,3,4,5,6,7,8,9,10,11,12,13,14,15);
            acc[t] = __builtin_amdgcn_wmma_f32_16x16x32_f16(false, af, false, bf,
                                                            (short)0, acc[t], false, false);
        }
    }
#pragma unroll
    for (int t = 0; t < NT; ++t) {
        const int col = (tn0 + t) * 16 + r;
#pragma unroll
        for (int j = 0; j < 8; ++j)
            C[(long)(tm * 16 + j + 8 * g) * ldc + col] = acc[t][j];
    }
}

// =====================================================================
// DFT basis tables (f16), generated directly in the layout the GEMM wants.
// which: 0=F fwd rows [64][128] (A-op)            a=cos  b=-sin
//        1=G fwd cols Bt [32kx][128x]             a=cos  b=-sin  c=+sin
//        2=E inv rows [128y][64j] (A-op)          a=cos  b=+sin  c=-sin
//        3=C inv cols Bt [128x][32kx] (fold 1/2)  a=w cos b=-w sin
//        4=E32 inv rows [32y][32j], j>=16 zero    a=cos  b=+sin  c=-sin
//        5=C8 inv cols Bt [32x][32kx], kx>=8 zero a=w cos b=-w sin
// =====================================================================
__global__ void fill_basis_k(half_t* a, half_t* b, half_t* c, int which, int n)
{
    int i = blockIdx.x * blockDim.x + threadIdx.x;
    if (i >= n) return;
    float av = 0.f, bv = 0.f, cv = 0.f;
    if (which == 0) {
        int row = i / 128, y = i % 128;
        int ky = (row < 32) ? row : row + 64;
        float t = 2.0f * PI_F * (float)ky * (float)y / 128.0f;
        av = cosf(t); bv = -sinf(t);
    } else if (which == 1) {
        int kx = i / 128, x = i % 128;
        float t = 2.0f * PI_F * (float)kx * (float)x / 128.0f;
        av = cosf(t); bv = -sinf(t); cv = sinf(t);
    } else if (which == 2) {
        int y = i / 64, j = i % 64;
        int ky = (j < 32) ? j : j + 64;
        float t = 2.0f * PI_F * (float)ky * (float)y / 128.0f;
        av = cosf(t); bv = sinf(t); cv = -sinf(t);
    } else if (which == 3) {
        int x = i / 32, kx = i % 32;
        float w = (kx == 0) ? 1.0f : 2.0f;
        float t = 2.0f * PI_F * (float)kx * (float)x / 128.0f;
        av = w * cosf(t); bv = -w * sinf(t);
    } else if (which == 4) {
        int y = i / 32, j = i % 32;
        if (j < 16) {
            int ky = (j < 8) ? j : j + 16;
            float t = 2.0f * PI_F * (float)ky * (float)y / 32.0f;
            av = cosf(t); bv = sinf(t); cv = -sinf(t);
        }
    } else {
        int x = i / 32, kx = i % 32;
        if (kx < 8) {
            float w = (kx == 0) ? 1.0f : 2.0f;
            float t = 2.0f * PI_F * (float)kx * (float)x / 32.0f;
            av = w * cosf(t); bv = -w * sinf(t);
        }
    }
    a[i] = (half_t)av; b[i] = (half_t)bv;
    if (c) c[i] = (half_t)cv;
}

// ---------------- f32 -> f16 ----------------
__global__ void cvt_k(const float* __restrict__ s, half_t* __restrict__ d, long n)
{
    long i = (long)blockIdx.x * blockDim.x + threadIdx.x;
    if (i < n) d[i] = (half_t)s[i];
}

// ---------------- f32 -> f16 with per-image 128x128 transpose ----------------
__global__ void cvt_t_k(const float* __restrict__ s, half_t* __restrict__ d, long n)
{
    long i = (long)blockIdx.x * blockDim.x + threadIdx.x;
    if (i >= n) return;
    long img = i >> 14;
    int  w   = (int)(i & 16383), y = w >> 7, x = w & 127;
    d[img * 16384 + (long)x * 128 + y] = (half_t)s[i];
}

// ---------------- v [8][64][16384] f32 -> vhT [8][16384][64] f16 ----------------
__global__ void cvt_vt_k(const float* __restrict__ s, half_t* __restrict__ d, long n)
{
    long i = (long)blockIdx.x * blockDim.x + threadIdx.x;
    if (i >= n) return;
    long img = i >> 14;                     // 0..511
    int  dd  = (int)(i & 16383);
    long b   = img >> 6;
    int  ss  = (int)(img & 63);
    d[b * 1048576 + (long)dd * 64 + ss] = (half_t)s[i];
}

// ---------------- zero fill ----------------
__global__ void zero_k(float* __restrict__ p, long n)
{
    long i = (long)blockIdx.x * blockDim.x + threadIdx.x;
    if (i < n) p[i] = 0.0f;
}

// ---------------- elementwise add ----------------
__global__ void add_k(const float* __restrict__ a, const float* __restrict__ b,
                      float* __restrict__ o, long n)
{
    long i = (long)blockIdx.x * blockDim.x + threadIdx.x;
    if (i < n) o[i] = a[i] + b[i];
}

// ---------------- instance norm; optional f32 out + f16 transposed out ----------------
__global__ void instnorm_k(const float* __restrict__ x, const float* gp, const float* bp,
                           float* out32, half_t* out16t, int n)
{
    const long img = blockIdx.x;
    const float* xi = x + img * (long)n;
    __shared__ float ss[256], sq[256];
    float s = 0.f, q = 0.f;
    for (int i = threadIdx.x; i < n; i += blockDim.x) { float v = xi[i]; s += v; q += v * v; }
    ss[threadIdx.x] = s; sq[threadIdx.x] = q; __syncthreads();
    for (int st = blockDim.x >> 1; st > 0; st >>= 1) {
        if (threadIdx.x < st) { ss[threadIdx.x] += ss[threadIdx.x + st]; sq[threadIdx.x] += sq[threadIdx.x + st]; }
        __syncthreads();
    }
    float mean = ss[0] / n;
    float var  = sq[0] / n - mean * mean;
    float inv  = rsqrtf(var + 1e-5f);
    float g = gp[0], b = bp[0];
    for (int i = threadIdx.x; i < n; i += blockDim.x) {
        float v = (xi[i] - mean) * inv * g + b;
        if (out32)  out32[img * (long)n + i] = v;
        if (out16t) out16t[img * (long)n + (long)(i & 127) * 128 + (i >> 7)] = (half_t)v;
    }
}

// ---------------- mixer tail: h = instnorm(Yc+bias)*g+b + sw*h ; opt GELU; f16 transposed ----
__global__ void mixer_post_k(const float* __restrict__ Yc, float* __restrict__ h,
                             half_t* __restrict__ h16t, const float* biasp, const float* swp,
                             const float* gp, const float* bp, int n, int do_gelu)
{
    const long img = blockIdx.x;
    const float* yi = Yc + img * (long)n;
    float bias = biasp[0];
    __shared__ float ss[256], sq[256];
    float s = 0.f, q = 0.f;
    for (int i = threadIdx.x; i < n; i += blockDim.x) { float v = yi[i] + bias; s += v; q += v * v; }
    ss[threadIdx.x] = s; sq[threadIdx.x] = q; __syncthreads();
    for (int st = blockDim.x >> 1; st > 0; st >>= 1) {
        if (threadIdx.x < st) { ss[threadIdx.x] += ss[threadIdx.x + st]; sq[threadIdx.x] += sq[threadIdx.x + st]; }
        __syncthreads();
    }
    float mean = ss[0] / n;
    float var  = sq[0] / n - mean * mean;
    float inv  = rsqrtf(var + 1e-5f);
    float g = gp[0], b = bp[0], sw = swp[0];
    for (int i = threadIdx.x; i < n; i += blockDim.x) {
        long idx = img * (long)n + i;
        float v = ((yi[i] + bias) - mean) * inv * g + b + sw * h[idx];
        if (do_gelu) v = 0.5f * v * (1.0f + erff(v * 0.70710678118654752f));
        h[idx] = v;
        h16t[img * (long)n + (long)(i & 127) * 128 + (i >> 7)] = (half_t)v;
    }
}

// ---------------- spectral multiply (m=32): O^T[kx][r] = (S*w/16384), f16 ----------------
__global__ void specmul_full_k(const float* __restrict__ Sr, const float* __restrict__ Si,
                               const float* __restrict__ w1r, const float* __restrict__ w1i,
                               const float* __restrict__ w2r, const float* __restrict__ w2i,
                               half_t* __restrict__ OrT, half_t* __restrict__ OiT, long n)
{
    long i = (long)blockIdx.x * blockDim.x + threadIdx.x;   // n = NB*2048, layout [img][kx][r]
    if (i >= n) return;
    int  r   = (int)(i & 63);
    int  kx  = (int)((i >> 6) & 31);
    long img = i >> 11;
    const float sc = 1.0f / 16384.0f;                        // rfft2 norm='forward'
    float wr, wi;
    if (r < 32) { wr = w1r[r * 32 + kx]; wi = w1i[r * 32 + kx]; }
    else        { wr = w2r[(r - 32) * 32 + kx]; wi = w2i[(r - 32) * 32 + kx]; }
    long src = img * 2048 + (long)r * 32 + kx;
    float sr = Sr[src], si = Si[src];
    OrT[i] = (half_t)((sr * wr - si * wi) * sc);
    OiT[i] = (half_t)((sr * wi + si * wr) * sc);
}

// ---------------- spectral multiply (K/Q, m=8): O^T [16kx][32j] zero-padded ----------------
__global__ void specmul_small_k(const float* __restrict__ Sr, const float* __restrict__ Si,
                                const float* __restrict__ w1r, const float* __restrict__ w1i,
                                const float* __restrict__ w2r, const float* __restrict__ w2i,
                                half_t* __restrict__ OrT, half_t* __restrict__ OiT, long n)
{
    long i = (long)blockIdx.x * blockDim.x + threadIdx.x;   // n = NB*512, layout [img][kx][j]
    if (i >= n) return;
    int  j   = (int)(i & 31);
    int  kx  = (int)((i >> 5) & 15);
    long img = i >> 9;
    if (kx >= 8 || j >= 16) { OrT[i] = (half_t)0.0f; OiT[i] = (half_t)0.0f; return; }
    const float sc = 1.0f / 16384.0f;
    int rsrc = (j < 8) ? j : j + 48;                 // spectrum rows {0..7, 56..63}
    float wr, wi;
    if (j < 8) { wr = w1r[j * 8 + kx]; wi = w1i[j * 8 + kx]; }
    else       { wr = w2r[(j - 8) * 8 + kx]; wi = w2i[(j - 8) * 8 + kx]; }
    long src = img * 2048 + (long)rsrc * 32 + kx;
    float sr = Sr[src], si = Si[src];
    OrT[i] = (half_t)((sr * wr - si * wi) * sc);
    OiT[i] = (half_t)((sr * wi + si * wr) * sc);
}

// ---------------- softmax over 64 cols -> f16 ----------------
__global__ void softmax_k(const float* __restrict__ S, half_t* __restrict__ P,
                          int T, float scale)
{
    const long row = blockIdx.x;
    const float* s = S + row * (long)T;
    __shared__ float red[64];
    float v = s[threadIdx.x] * scale;
    red[threadIdx.x] = v; __syncthreads();
    for (int st = 32; st > 0; st >>= 1) {
        if (threadIdx.x < st) red[threadIdx.x] = fmaxf(red[threadIdx.x], red[threadIdx.x + st]);
        __syncthreads();
    }
    float m = red[0]; __syncthreads();
    float e = expf(v - m);
    red[threadIdx.x] = e; __syncthreads();
    for (int st = 32; st > 0; st >>= 1) {
        if (threadIdx.x < st) red[threadIdx.x] += red[threadIdx.x + st];
        __syncthreads();
    }
    P[row * (long)T + threadIdx.x] = (half_t)(e / red[0]);
}

// =====================================================================
extern "C" void kernel_launch(void* const* d_in, const int* in_sizes, int n_in,
                              void* d_out, int out_size, void* d_ws, size_t ws_size,
                              hipStream_t stream)
{
    (void)in_sizes; (void)n_in; (void)out_size; (void)ws_size;
    const int  NB   = 512;
    const long IMG  = 16384;
    const long NTOT = (long)NB * IMG;

    const float* x   = (const float*)d_in[0];
    const float* n1g = (const float*)d_in[1];
    const float* n1b = (const float*)d_in[2];
    const float* Kw[4]  = {(const float*)d_in[3],(const float*)d_in[4],(const float*)d_in[5],(const float*)d_in[6]};
    const float* Qw[4]  = {(const float*)d_in[7],(const float*)d_in[8],(const float*)d_in[9],(const float*)d_in[10]};
    const float* Vw[4]  = {(const float*)d_in[11],(const float*)d_in[12],(const float*)d_in[13],(const float*)d_in[14]};
    const float* Pw[4]  = {(const float*)d_in[15],(const float*)d_in[16],(const float*)d_in[17],(const float*)d_in[18]};
    const float* ang = (const float*)d_in[19];
    const float* anb = (const float*)d_in[20];
    const float* L0w[4] = {(const float*)d_in[21],(const float*)d_in[22],(const float*)d_in[23],(const float*)d_in[24]};
    const float* L0bias = (const float*)d_in[25];
    const float* L0sw   = (const float*)d_in[26];
    const float* L0ng   = (const float*)d_in[27];
    const float* L0nb   = (const float*)d_in[28];
    const float* L1w[4] = {(const float*)d_in[29],(const float*)d_in[30],(const float*)d_in[31],(const float*)d_in[32]};
    const float* L1bias = (const float*)d_in[33];
    const float* L1sw   = (const float*)d_in[34];
    const float* L1ng   = (const float*)d_in[35];
    const float* L1nb   = (const float*)d_in[36];
    float* out = (float*)d_out;

    char* base = (char*)d_ws;
    size_t off = 0;
    auto alloc = [&](size_t bytes) -> char* {
        off = (off + 255) & ~(size_t)255;
        char* p = base + off; off += bytes; return p;
    };
    // basis tables
    half_t* Fr   = (half_t*)alloc(64 * 128 * 2);
    half_t* Fi   = (half_t*)alloc(64 * 128 * 2);
    half_t* GrT  = (half_t*)alloc(32 * 128 * 2);
    half_t* GiT  = (half_t*)alloc(32 * 128 * 2);
    half_t* GinT = (half_t*)alloc(32 * 128 * 2);
    half_t* Er   = (half_t*)alloc(128 * 64 * 2);
    half_t* Ei   = (half_t*)alloc(128 * 64 * 2);
    half_t* Ein  = (half_t*)alloc(128 * 64 * 2);
    half_t* CcT  = (half_t*)alloc(128 * 32 * 2);
    half_t* CsT  = (half_t*)alloc(128 * 32 * 2);
    half_t* E32r = (half_t*)alloc(32 * 32 * 2);
    half_t* E32i = (half_t*)alloc(32 * 32 * 2);
    half_t* E32in= (half_t*)alloc(32 * 32 * 2);
    half_t* C8cT = (half_t*)alloc(32 * 32 * 2);
    half_t* C8sT = (half_t*)alloc(32 * 32 * 2);
    // big intermediates
    half_t* img16 = (half_t*)alloc(NTOT * 2);                 // f16 images, TRANSPOSED [x][y]
    float*  Arf = (float*)alloc((long)NB * 8192 * 4);
    float*  Aif = (float*)alloc((long)NB * 8192 * 4);
    half_t* Arh = (half_t*)alloc((long)NB * 8192 * 2);
    half_t* Aih = (half_t*)alloc((long)NB * 8192 * 2);
    float*  Srf = (float*)alloc((long)NB * 2048 * 4);
    float*  Sif = (float*)alloc((long)NB * 2048 * 4);
    half_t* OrT = (half_t*)alloc((long)NB * 2048 * 2);
    half_t* OiT = (half_t*)alloc((long)NB * 2048 * 2);
    half_t* OKr = (half_t*)alloc((long)NB * 512 * 2);
    half_t* OKi = (half_t*)alloc((long)NB * 512 * 2);
    half_t* OQr = (half_t*)alloc((long)NB * 512 * 2);
    half_t* OQi = (half_t*)alloc((long)NB * 512 * 2);
    float*  Zrf = (float*)alloc((long)NB * 4096 * 4);
    float*  Zif = (float*)alloc((long)NB * 4096 * 4);
    half_t* Zrh = (half_t*)alloc((long)NB * 4096 * 2);
    half_t* Zih = (half_t*)alloc((long)NB * 4096 * 2);
    float*  Y   = (float*)alloc(NTOT * 4);                    // conv out / atten
    float*  Y2  = (float*)alloc(NTOT * 4);                    // atten residual
    float*  Hb  = (float*)alloc(NTOT * 4);                    // mixer hidden f32
    half_t* vhT = (half_t*)alloc(NTOT * 2);                   // [8][16384][64]
    float*  qf  = (float*)alloc((long)NB * 1024 * 4);
    float*  kf  = (float*)alloc((long)NB * 1024 * 4);
    half_t* qh  = (half_t*)alloc((long)NB * 1024 * 2);
    half_t* kh  = (half_t*)alloc((long)NB * 1024 * 2);
    float*  Sc  = (float*)alloc(8 * 64 * 64 * 4);
    half_t* Pp  = (half_t*)alloc(8 * 64 * 64 * 2);

    // dispatch largest NT in {4,2,1} dividing tilesN -> zero conditionals in-loop
    auto gemm = [&](const half_t* A, const half_t* Bt, float* C, int M, int N, int K,
                    int lda, int ldb, int ldc, long sA, long sB, long sC, int beta, int batch) {
        int tilesN = N >> 4;
        int nt = (tilesN % 4 == 0) ? 4 : (tilesN % 2 == 0) ? 2 : 1;
        int tiles = (M >> 4) * (tilesN / nt);
        dim3 blk(32, 8), grd((tiles + 7) / 8, batch);
        if (nt == 4)
            gemm16_k<4><<<grd, blk, 0, stream>>>(A, Bt, C, M, N, K, lda, ldb, ldc, sA, sB, sC, beta);
        else if (nt == 2)
            gemm16_k<2><<<grd, blk, 0, stream>>>(A, Bt, C, M, N, K, lda, ldb, ldc, sA, sB, sC, beta);
        else
            gemm16_k<1><<<grd, blk, 0, stream>>>(A, Bt, C, M, N, K, lda, ldb, ldc, sA, sB, sC, beta);
    };
    auto cvt = [&](const float* s, half_t* d, long n) {
        cvt_k<<<(unsigned)((n + 255) / 256), 256, 0, stream>>>(s, d, n);
    };
    auto fill = [&](half_t* a, half_t* b, half_t* c, int which, int n) {
        fill_basis_k<<<(n + 255) / 256, 256, 0, stream>>>(a, b, c, which, n);
    };
    // truncated forward DFT of img16 (transposed f16 images) -> Srf/Sif [NB][64][32]
    auto forward = [&]() {
        gemm(Fr, img16, Arf, 64, 128, 128, 128, 128, 128, 0, IMG, 8192, 0, NB);
        gemm(Fi, img16, Aif, 64, 128, 128, 128, 128, 128, 0, IMG, 8192, 0, NB);
        cvt(Arf, Arh, (long)NB * 8192); cvt(Aif, Aih, (long)NB * 8192);
        gemm(Arh, GrT,  Srf, 64, 32, 128, 128, 128, 32, 8192, 0, 2048, 0, NB);
        gemm(Aih, GinT, Srf, 64, 32, 128, 128, 128, 32, 8192, 0, 2048, 1, NB);
        gemm(Arh, GiT,  Sif, 64, 32, 128, 128, 128, 32, 8192, 0, 2048, 0, NB);
        gemm(Aih, GrT,  Sif, 64, 32, 128, 128, 128, 32, 8192, 0, 2048, 1, NB);
    };
    // inverse 128x128 DFT of OrT/OiT -> Yout f32
    auto invFull = [&](float* Yout) {
        gemm(Er,  OrT, Zrf, 128, 32, 64, 64, 64, 32, 0, 2048, 4096, 0, NB);
        gemm(Ein, OiT, Zrf, 128, 32, 64, 64, 64, 32, 0, 2048, 4096, 1, NB);
        gemm(Er,  OiT, Zif, 128, 32, 64, 64, 64, 32, 0, 2048, 4096, 0, NB);
        gemm(Ei,  OrT, Zif, 128, 32, 64, 64, 64, 32, 0, 2048, 4096, 1, NB);
        cvt(Zrf, Zrh, (long)NB * 4096); cvt(Zif, Zih, (long)NB * 4096);
        gemm(Zrh, CcT, Yout, 128, 128, 32, 32, 32, 128, 4096, 0, IMG, 0, NB);
        gemm(Zih, CsT, Yout, 128, 128, 32, 32, 32, 128, 4096, 0, IMG, 1, NB);
    };
    // inverse 32x32 DFT (K/Q) of padded mode tiles -> o f32 [NB][32][32]
    auto invSmall = [&](const half_t* ORe, const half_t* OIm, float* o) {
        gemm(E32r,  ORe, Zrf, 32, 16, 32, 32, 32, 32, 0, 512, 1024, 0, NB);
        gemm(E32in, OIm, Zrf, 32, 16, 32, 32, 32, 32, 0, 512, 1024, 1, NB);
        gemm(E32r,  OIm, Zif, 32, 16, 32, 32, 32, 32, 0, 512, 1024, 0, NB);
        gemm(E32i,  ORe, Zif, 32, 16, 32, 32, 32, 32, 0, 512, 1024, 1, NB);
        cvt(Zrf, Zrh, (long)NB * 1024); cvt(Zif, Zih, (long)NB * 1024);
        gemm(Zrh, C8cT, o, 32, 32, 32, 32, 32, 32, 1024, 0, 1024, 0, NB);
        gemm(Zih, C8sT, o, 32, 32, 32, 32, 32, 32, 1024, 0, 1024, 1, NB);
    };
    auto specF = [&](const float* const w[4]) {
        long n = (long)NB * 2048;
        specmul_full_k<<<(unsigned)((n + 255) / 256), 256, 0, stream>>>(
            Srf, Sif, w[0], w[1], w[2], w[3], OrT, OiT, n);
    };

    // ---- basis tables ----
    fill(Fr,   Fi,   nullptr, 0, 64 * 128);
    fill(GrT,  GiT,  GinT,    1, 32 * 128);
    fill(Er,   Ei,   Ein,     2, 128 * 64);
    fill(CcT,  CsT,  nullptr, 3, 128 * 32);
    fill(E32r, E32i, E32in,   4, 32 * 32);
    fill(C8cT, C8sT, nullptr, 5, 32 * 32);
    // zero the padded halves of the small-Z buffers (cols 16..31 never written)
    zero_k<<<(unsigned)(((long)NB * 1024 + 255) / 256), 256, 0, stream>>>(Zrf, (long)NB * 1024);
    zero_k<<<(unsigned)(((long)NB * 1024 + 255) / 256), 256, 0, stream>>>(Zif, (long)NB * 1024);

    // ---- 1) norm1 -> transposed f16 images ----
    instnorm_k<<<NB, 256, 0, stream>>>(x, n1g, n1b, nullptr, img16, (int)IMG);

    // ---- 2) shared forward spectrum ----
    forward();

    // ---- 3) K/Q/V spectral multiplies ----
    {
        long n = (long)NB * 512;
        specmul_small_k<<<(unsigned)((n + 255) / 256), 256, 0, stream>>>(
            Srf, Sif, Kw[0], Kw[1], Kw[2], Kw[3], OKr, OKi, n);
        specmul_small_k<<<(unsigned)((n + 255) / 256), 256, 0, stream>>>(
            Srf, Sif, Qw[0], Qw[1], Qw[2], Qw[3], OQr, OQi, n);
    }
    specF(Vw);

    // ---- 4) inverse DFTs ----
    invSmall(OKr, OKi, kf);
    invSmall(OQr, OQi, qf);
    cvt(kf, kh, (long)NB * 1024);
    cvt(qf, qh, (long)NB * 1024);
    invFull(Y);                                   // Y = v
    cvt_vt_k<<<(unsigned)((NTOT + 255) / 256), 256, 0, stream>>>(Y, vhT, NTOT);

    // ---- 5) attention ----
    gemm(qh, kh, Sc, 64, 64, 1024, 1024, 1024, 64, 65536, 65536, 4096, 0, 8);
    softmax_k<<<512, 64, 0, stream>>>(Sc, Pp, 64, 1.0f / 1024.0f);
    gemm(Pp, vhT, Y, 64, 16384, 64, 64, 64, 16384, 4096, 1048576, 1048576, 0, 8);  // Y = atten

    // ---- 6) proj conv + residual ----
    cvt_t_k<<<(unsigned)((NTOT + 255) / 256), 256, 0, stream>>>(Y, img16, NTOT);
    forward();
    specF(Pw);
    invFull(Y2);
    add_k<<<(unsigned)((NTOT + 255) / 256), 256, 0, stream>>>(Y2, x, Y2, NTOT);    // atten += xa

    // ---- 7) attn_norm ----
    instnorm_k<<<NB, 256, 0, stream>>>(Y2, ang, anb, Hb, img16, (int)IMG);

    // ---- 8) mixer layer0 (gelu) ----
    forward();
    specF(L0w);
    invFull(Y);
    mixer_post_k<<<NB, 256, 0, stream>>>(Y, Hb, img16, L0bias, L0sw, L0ng, L0nb, (int)IMG, 1);

    // ---- 9) mixer layer1 ----
    forward();
    specF(L1w);
    invFull(Y);
    mixer_post_k<<<NB, 256, 0, stream>>>(Y, Hb, img16, L1bias, L1sw, L1ng, L1nb, (int)IMG, 0);

    // ---- 10) out = h + atten ----
    add_k<<<(unsigned)((NTOT + 255) / 256), 256, 0, stream>>>(Hb, Y2, out, NTOT);
}
